// Mamba_4406636446264
// MI455X (gfx1250) — matrix operand order repaired and checked by reading, not compiled, not run on hardware
//
#include <hip/hip_runtime.h>

typedef __attribute__((ext_vector_type(16))) __bf16 v16bf;
typedef __attribute__((ext_vector_type(8)))  __bf16 v8bf;
typedef __attribute__((ext_vector_type(8)))  float  v8f;
typedef __attribute__((ext_vector_type(4)))  unsigned int u32x4;
typedef __attribute__((ext_vector_type(8)))  unsigned int u32x8;

#define E_   512
#define D_   128
#define L_   512
#define B_   2
#define V_   32000
#define DT_  8
#define KC_  4
#define TOK  (B_*L_)

__device__ __forceinline__ float silu_f(float x) { return x / (1.0f + __expf(-x)); }

// ---- per-lane async copy global->LDS (ASYNCcnt path), 16B granule ----
__device__ __forceinline__ void async_ld_b128(unsigned lds_off, unsigned long long gaddr) {
    asm volatile("global_load_async_to_lds_b128 %0, %1, off"
                 :: "v"(lds_off), "v"(gaddr) : "memory");
}
__device__ __forceinline__ void async_ld_b128_o16(unsigned lds_off, unsigned long long gaddr) {
    // INST_OFFSET applies to BOTH the LDS and the global address (ISA 10.x pseudocode)
    asm volatile("global_load_async_to_lds_b128 %0, %1, off offset:16"
                 :: "v"(lds_off), "v"(gaddr) : "memory");
}
__device__ __forceinline__ void wait_asynccnt0() {
    asm volatile("s_wait_asynccnt 0x0" ::: "memory");
}

// ---- Tensor Data Mover: 2-D tile load (fp32 elements) via D# in SGPRs ----
__device__ __forceinline__ void tdm_load_2d_f32(unsigned lds_addr, const void* gptr,
                                                unsigned tile_d0, unsigned tile_d1,
                                                unsigned tensor_d0, unsigned tensor_d1,
                                                unsigned stride0) {
    unsigned long long ga = (unsigned long long)(size_t)gptr;
    u32x4 g0;
    g0[0] = 1u;                                                   // count=1, user desc
    g0[1] = lds_addr;                                             // LDS byte address
    g0[2] = (unsigned)ga;                                         // global_addr[31:0]
    g0[3] = (unsigned)((ga >> 32) & 0x01ffffffu) | 0x80000000u;   // addr[56:32] | type=2
    u32x8 g1;
    g1[0] = (2u << 16);                                           // data_size = 4B
    g1[1] = (tensor_d0 & 0xffffu) << 16;                          // tensor_dim0[15:0]
    g1[2] = (tensor_d0 >> 16) | ((tensor_d1 & 0xffffu) << 16);    // dim0 hi | dim1 lo
    g1[3] = (tensor_d1 >> 16) | (tile_d0 << 16);                  // dim1 hi | tile_dim0
    g1[4] = tile_d1;                                              // tile_dim1 (tile_dim2=0)
    g1[5] = stride0;                                              // tensor_dim0_stride lo32
    g1[6] = 0u;
    g1[7] = 0u;
    asm volatile("tensor_load_to_lds %0, %1" :: "s"(g0), "s"(g1) : "memory");
}

// ---------------- fp32 -> bf16 convert ----------------
__global__ void k_cvt_bf16(const float* __restrict__ in, __bf16* __restrict__ out, int n) {
    for (int i = blockIdx.x * blockDim.x + threadIdx.x; i < n; i += gridDim.x * blockDim.x)
        out[i] = (__bf16)in[i];
}

// ---------------- A = -exp(A_log) ----------------
__global__ void k_amat(const float* __restrict__ alog, float* __restrict__ A, int n) {
    for (int i = blockIdx.x * blockDim.x + threadIdx.x; i < n; i += gridDim.x * blockDim.x)
        A[i] = -__expf(alog[i]);
}

// ---------------- embedding gather: 1 block / token ----------------
__global__ void k_embed(const int* __restrict__ x, const float* __restrict__ emb,
                        float* __restrict__ h) {
    int tok = blockIdx.x;
    int row = x[tok];
    const float4* src = (const float4*)(emb + (size_t)row * E_);
    float4* dst = (float4*)(h + (size_t)tok * E_);
    dst[threadIdx.x] = src[threadIdx.x];   // 128 threads * float4 = 512
}

// ---------------- RMS norm over E, emit bf16 ----------------
__global__ void k_rms(const float* __restrict__ h, const float* __restrict__ w,
                      __bf16* __restrict__ out) {
    __shared__ float red[256];
    int tok = blockIdx.x, t = threadIdx.x;
    const float* row = h + (size_t)tok * E_;
    float v0 = row[t], v1 = row[t + 256];
    red[t] = v0 * v0 + v1 * v1;
    __syncthreads();
    for (int off = 128; off > 0; off >>= 1) {
        if (t < off) red[t] += red[t + off];
        __syncthreads();
    }
    float rs = rsqrtf(red[0] * (1.0f / E_) + 1e-5f);
    out[(size_t)tok * E_ + t]       = (__bf16)(v0 * rs * w[t]);
    out[(size_t)tok * E_ + t + 256] = (__bf16)(v1 * rs * w[t + 256]);
}

// ---------------- depthwise conv (VALID, K=4) + bias + SiLU, zero-pad tail ----------------
__global__ void k_conv(const float* __restrict__ xl, const float* __restrict__ cw,
                       const float* __restrict__ cb, float* __restrict__ xc,
                       __bf16* __restrict__ xcb) {
    int blk = blockIdx.x;
    int b = blk / L_, t = blk % L_, d = threadIdx.x;
    float v = 0.0f;
    if (t <= L_ - KC_) {
        float s = cb[d];
#pragma unroll
        for (int k = 0; k < KC_; ++k)
            s = fmaf(xl[((size_t)(b * L_ + t + k)) * D_ + d], cw[d * KC_ + k], s);
        v = silu_f(s);
    }
    size_t o = (size_t)blk * D_ + d;
    xc[o] = v;
    xcb[o] = (__bf16)v;
}

// ---------------- delta = softplus((xc@Wdi^T+bdi)@Wd^T+bd), 1 block / token ----------------
__global__ void k_delta(const float* __restrict__ xc, const float* __restrict__ Wdi,
                        const float* __restrict__ bdi, const float* __restrict__ Wd,
                        const float* __restrict__ bd, float* __restrict__ delta) {
    __shared__ float xrow[D_];
    __shared__ float di[DT_];
    int tok = blockIdx.x, t = threadIdx.x;
    xrow[t] = xc[(size_t)tok * D_ + t];
    __syncthreads();
    if (t < DT_) {
        float s = bdi[t];
        for (int d = 0; d < D_; ++d) s = fmaf(xrow[d], Wdi[t * D_ + d], s);
        di[t] = s;
    }
    __syncthreads();
    float s = bd[t];
#pragma unroll
    for (int j = 0; j < DT_; ++j) s = fmaf(di[j], Wd[t * DT_ + j], s);
    float sp = (s > 20.0f) ? s : log1pf(__expf(s));
    delta[(size_t)tok * D_ + t] = sp;
}

// ---------------- per-token triangular solve: M = inv(dA) @ (exp(dA)-I); B_hat = M*dB*x ----
__global__ __launch_bounds__(128)
void k_bhat(const float* __restrict__ A, const float* __restrict__ delta,
            const float* __restrict__ Bm, const float* __restrict__ xc,
            float* __restrict__ Bhat) {
    __shared__ float M[D_ * D_];    // 64 KB of the WGP's 320 KB LDS
    int tok = blockIdx.x, n = threadIdx.x;
    float Bn = Bm[(size_t)tok * D_ + n];
    for (int d = 0; d < D_; ++d) {
        float dl = delta[(size_t)tok * D_ + d];      // uniform broadcast
        float acc = 0.0f;
        for (int j = 0; j < d; ++j)
            acc = fmaf(dl * A[d * D_ + j], M[j * D_ + n], acc);
        float dAn = dl * A[d * D_ + n];
        float rhs = __expf(dAn) - ((n == d) ? 1.0f : 0.0f);
        float Mdn = (rhs - acc) / (dl * A[d * D_ + d]);
        M[d * D_ + n] = Mdn;
        Bhat[((size_t)tok * D_ + d) * D_ + n] = Mdn * (dl * Bn) * xc[(size_t)tok * D_ + d];
        __syncthreads();
    }
}

// ---------------- sequential scan over L ----------------
// TDM pre-gathers the three strided columns delta[:,d], xc[:,d], xr[:,d] (stride 512B)
// into contiguous LDS, then the L-loop runs from LDS broadcasts.
__global__ __launch_bounds__(128)
void k_scan(const float* __restrict__ A, const float* __restrict__ delta,
            const float* __restrict__ Bhat, const float* __restrict__ Cm,
            const float* __restrict__ xc, const float* __restrict__ Dp,
            const float* __restrict__ xr, __bf16* __restrict__ ybf) {
    __shared__ float red[D_];
    __shared__ float dCol[L_];
    __shared__ float xcCol[L_];
    __shared__ float xrCol[L_];
    int b = blockIdx.x / D_, d = blockIdx.x % D_, n = threadIdx.x;
    size_t base = (size_t)b * L_ * D_ + d;          // element offset of column start

    if (n < 32) {   // wave 0 issues the tensor DMAs (TENSORcnt), EXEC ignored by TDM
        tdm_load_2d_f32((unsigned)(size_t)&dCol[0],  delta + base, 1u, (unsigned)L_,
                        (unsigned)D_, (unsigned)L_, (unsigned)D_);
        tdm_load_2d_f32((unsigned)(size_t)&xcCol[0], xc + base,    1u, (unsigned)L_,
                        (unsigned)D_, (unsigned)L_, (unsigned)D_);
        tdm_load_2d_f32((unsigned)(size_t)&xrCol[0], xr + base,    1u, (unsigned)L_,
                        (unsigned)D_, (unsigned)L_, (unsigned)D_);
        __builtin_amdgcn_s_wait_tensorcnt(0);
    }
    __syncthreads();

    float An = A[d * D_ + n];
    float dpd = Dp[d];
    float k = 0.0f;
    for (int l = 0; l < L_; ++l) {
        size_t tok = (size_t)b * L_ + l;
        if (l + 1 < L_)
            __builtin_prefetch(Bhat + ((tok + 1) * D_ + d) * D_ + n, 0, 1);
        float dl = dCol[l];
        float ah = __expf(dl * An);
        k = fmaf(ah, k, Bhat[(tok * D_ + d) * D_ + n]);
        red[n] = k * Cm[tok * D_ + n];
        __syncthreads();
        for (int off = 64; off > 0; off >>= 1) {
            if (n < off) red[n] += red[n + off];
            __syncthreads();
        }
        if (n == 0) {
            float y = red[0] + xcCol[l] * dpd;
            ybf[tok * D_ + d] = (__bf16)(y * xrCol[l]);
        }
        __syncthreads();
    }
}

// ---------------- bf16 WMMA GEMM: C[M,N] = A[M,K] * Bw[N,K]^T (+bias,+act,+resid) ----------
// 128x128 tile, 256 threads = 8 waves (4 M-waves x 2 N-waves), 2x4 16x16 sub-tiles / wave.
// Double-buffered LDS fed by global_load_async_to_lds_b128 (ASYNCcnt), next tile issued
// before computing the current one; one barrier per K-step.
#define BM  128
#define BN  128
#define BK  32
#define LDT 40   // padded LDS row stride (bf16 elems); 80B rows keep 16B alignment

__global__ __launch_bounds__(256)
void k_gemm_bf16(const __bf16* __restrict__ A, const __bf16* __restrict__ Bw,
                 const float* __restrict__ bias, const float* __restrict__ resid,
                 float* __restrict__ outf, __bf16* __restrict__ outb,
                 int M, int N, int Kd, int act) {
    __shared__ __bf16 As[2][BM * LDT];
    __shared__ __bf16 Bs[2][BN * LDT];
    int tid  = threadIdx.x;
    int lane = tid & 31, wid = tid >> 5;
    int wn = wid & 1, wm = wid >> 1;             // 2 N-waves, 4 M-waves
    int mblock = blockIdx.y * BM, nblock = blockIdx.x * BN;
    int lrow = tid >> 1;                          // 0..127: tile row loaded by this thread
    int lk   = (tid & 1) * 16;                    // 0 or 16 within 32-wide K chunk
    int hi = lane >> 4, lo16 = lane & 15;

    unsigned ldsA0 = (unsigned)(size_t)&As[0][lrow * LDT + lk];
    unsigned ldsB0 = (unsigned)(size_t)&Bs[0][lrow * LDT + lk];
    const unsigned bufStride = (unsigned)(BM * LDT * sizeof(__bf16));
    const __bf16* gA0 = A  + (size_t)(mblock + lrow) * Kd + lk;
    const __bf16* gB0 = Bw + (size_t)(nblock + lrow) * Kd + lk;

    auto issue_tile = [&](int buf, int k0) {
        unsigned la = ldsA0 + (unsigned)buf * bufStride;
        unsigned lb = ldsB0 + (unsigned)buf * bufStride;
        unsigned long long pa = (unsigned long long)(size_t)(gA0 + k0);
        unsigned long long pb = (unsigned long long)(size_t)(gB0 + k0);
        async_ld_b128(la, pa);
        async_ld_b128_o16(la, pa);   // +16B on both LDS and global sides
        async_ld_b128(lb, pb);
        async_ld_b128_o16(lb, pb);
    };

    v8f acc[2][4];
#pragma unroll
    for (int i = 0; i < 2; ++i)
#pragma unroll
        for (int j = 0; j < 4; ++j)
#pragma unroll
            for (int v = 0; v < 8; ++v) acc[i][j][v] = 0.0f;

    int nTiles = Kd / BK;
    issue_tile(0, 0);
    for (int it = 0; it < nTiles; ++it) {
        wait_asynccnt0();            // my async ops (tile 'it') have landed in LDS
        __syncthreads();             // everyone's landed; prior compute done reading LDS
        if (it + 1 < nTiles) issue_tile((it + 1) & 1, (it + 1) * BK);

        const __bf16* Ab = As[it & 1];
        const __bf16* Bb = Bs[it & 1];

        // A fragments: lane holds row lane%16; K-runs [8*hi, +8) and [16+8*hi, +8)
        v16bf afr[2];
#pragma unroll
        for (int mt = 0; mt < 2; ++mt) {
            int mrow = wm * 32 + mt * 16 + lo16;
            v8bf aL = *(const v8bf*)(&Ab[mrow * LDT + hi * 8]);
            v8bf aH = *(const v8bf*)(&Ab[mrow * LDT + hi * 8 + 16]);
            afr[mt] = __builtin_shufflevector(aL, aH, 0,1,2,3,4,5,6,7,8,9,10,11,12,13,14,15);
        }
        // B fragments: lane holds col lane%16; K-runs [8*hi, +8) and [16+8*hi, +8)
#pragma unroll
        for (int nt = 0; nt < 4; ++nt) {
            int nrow = wn * 64 + nt * 16 + lo16;
            v8bf bL = *(const v8bf*)(&Bb[nrow * LDT + hi * 8]);
            v8bf bH = *(const v8bf*)(&Bb[nrow * LDT + hi * 8 + 16]);
            v16bf bfr = __builtin_shufflevector(bL, bH, 0,1,2,3,4,5,6,7,8,9,10,11,12,13,14,15);
#pragma unroll
            for (int mt = 0; mt < 2; ++mt)
                acc[mt][nt] = __builtin_amdgcn_wmma_f32_16x16x32_bf16(
                    false, afr[mt], false, bfr, (short)0, acc[mt][nt], false, false);
        }
    }

    // epilogue: C element (v, lane) -> M = v + 8*hi, N = lane%16
#pragma unroll
    for (int mt = 0; mt < 2; ++mt) {
        int row0 = mblock + wm * 32 + mt * 16 + hi * 8;
#pragma unroll
        for (int nt = 0; nt < 4; ++nt) {
            int col = nblock + wn * 64 + nt * 16 + lo16;
            float bv = bias ? bias[col] : 0.0f;
#pragma unroll
            for (int v = 0; v < 8; ++v) {
                int row = row0 + v;
                float val = acc[mt][nt][v] + bv;
                if (act == 1) val = silu_f(val);
                size_t o = (size_t)row * N + col;
                if (resid) val += resid[o];
                if (outf) outf[o] = val;
                if (outb) outb[o] = (__bf16)val;
            }
        }
    }
}

// ====================================================================================
extern "C" void kernel_launch(void* const* d_in, const int* in_sizes, int n_in,
                              void* d_out, int out_size, void* d_ws, size_t ws_size,
                              hipStream_t stream) {
    (void)in_sizes; (void)n_in; (void)out_size; (void)ws_size;

    const int*   x      = (const int*)d_in[0];
    const float* emb    = (const float*)d_in[1];
    const float* norm_w = (const float*)d_in[2 + 2 * 19 + 0];
    const float* head_w = (const float*)d_in[2 + 2 * 19 + 1];
    const float* head_b = (const float*)d_in[2 + 2 * 19 + 2];

    char* ws = (char*)d_ws;
    size_t off = 0;
    auto alloc = [&](size_t bytes) -> void* {
        void* p = ws + off;
        off += (bytes + 255) & ~(size_t)255;
        return p;
    };

    float*  hA      = (float*) alloc((size_t)TOK * E_ * 4);
    float*  hB      = (float*) alloc((size_t)TOK * E_ * 4);
    __bf16* xi_bf   = (__bf16*)alloc((size_t)TOK * E_ * 2);
    float*  xl_f    = (float*) alloc((size_t)TOK * D_ * 4);
    float*  xr_f    = (float*) alloc((size_t)TOK * D_ * 4);
    float*  xc_f    = (float*) alloc((size_t)TOK * D_ * 4);
    __bf16* xc_bf   = (__bf16*)alloc((size_t)TOK * D_ * 2);
    float*  Bm_f    = (float*) alloc((size_t)TOK * D_ * 4);
    float*  Cm_f    = (float*) alloc((size_t)TOK * D_ * 4);
    float*  delta_f = (float*) alloc((size_t)TOK * D_ * 4);
    __bf16* y_bf    = (__bf16*)alloc((size_t)TOK * D_ * 2);
    float*  Amat    = (float*) alloc((size_t)D_ * D_ * 4);
    __bf16* wl_bf   = (__bf16*)alloc((size_t)D_ * E_ * 2);
    __bf16* wr_bf   = (__bf16*)alloc((size_t)D_ * E_ * 2);
    __bf16* wb_bf   = (__bf16*)alloc((size_t)D_ * D_ * 2);
    __bf16* wc_bf   = (__bf16*)alloc((size_t)D_ * D_ * 2);
    __bf16* wo_bf   = (__bf16*)alloc((size_t)E_ * D_ * 2);
    __bf16* hn_bf   = (__bf16*)alloc((size_t)TOK * E_ * 2);
    __bf16* hw_bf   = (__bf16*)alloc((size_t)V_ * E_ * 2);
    float*  Bhat    = (float*) alloc((size_t)TOK * D_ * D_ * 4);   // 64 MB

    k_embed<<<TOK, 128, 0, stream>>>(x, emb, hA);

    float* hin = hA;
    float* hout = hB;
    for (int layer = 0; layer < 2; ++layer) {
        const float* const* p = (const float* const*)(d_in + 2 + layer * 19);
        const float *rms_w = p[0], *Wl = p[1], *bl = p[2], *Wr = p[3], *br = p[4];
        const float *conv_w = p[5], *conv_b = p[6], *A_log = p[7], *Dp = p[8];
        const float *Wb = p[9], *bb = p[10], *Wc = p[11], *bc = p[12];
        const float *Wdi = p[13], *bdi = p[14], *Wd = p[15], *bd = p[16];
        const float *Wo = p[17], *bo = p[18];

        k_cvt_bf16<<<256, 256, 0, stream>>>(Wl, wl_bf, D_ * E_);
        k_cvt_bf16<<<256, 256, 0, stream>>>(Wr, wr_bf, D_ * E_);
        k_cvt_bf16<<<64,  256, 0, stream>>>(Wb, wb_bf, D_ * D_);
        k_cvt_bf16<<<64,  256, 0, stream>>>(Wc, wc_bf, D_ * D_);
        k_cvt_bf16<<<256, 256, 0, stream>>>(Wo, wo_bf, E_ * D_);
        k_amat<<<64, 256, 0, stream>>>(A_log, Amat, D_ * D_);

        k_rms<<<TOK, 256, 0, stream>>>(hin, rms_w, xi_bf);
        k_gemm_bf16<<<dim3(D_ / 128, TOK / 128), 256, 0, stream>>>(
            xi_bf, wl_bf, bl, nullptr, xl_f, nullptr, TOK, D_, E_, 0);
        k_gemm_bf16<<<dim3(D_ / 128, TOK / 128), 256, 0, stream>>>(
            xi_bf, wr_bf, br, nullptr, xr_f, nullptr, TOK, D_, E_, 1);
        k_conv<<<TOK, D_, 0, stream>>>(xl_f, conv_w, conv_b, xc_f, xc_bf);
        k_gemm_bf16<<<dim3(D_ / 128, TOK / 128), 256, 0, stream>>>(
            xc_bf, wb_bf, bb, nullptr, Bm_f, nullptr, TOK, D_, D_, 0);
        k_gemm_bf16<<<dim3(D_ / 128, TOK / 128), 256, 0, stream>>>(
            xc_bf, wc_bf, bc, nullptr, Cm_f, nullptr, TOK, D_, D_, 0);
        k_delta<<<TOK, D_, 0, stream>>>(xc_f, Wdi, bdi, Wd, bd, delta_f);
        k_bhat<<<TOK, D_, 0, stream>>>(Amat, delta_f, Bm_f, xc_f, Bhat);
        k_scan<<<B_ * D_, D_, 0, stream>>>(Amat, delta_f, Bhat, Cm_f, xc_f, Dp, xr_f, y_bf);
        k_gemm_bf16<<<dim3(E_ / 128, TOK / 128), 256, 0, stream>>>(
            y_bf, wo_bf, bo, hin, hout, nullptr, TOK, E_, D_, 0);

        float* tmp = hin; hin = hout; hout = tmp;
    }

    k_rms<<<TOK, 256, 0, stream>>>(hin, norm_w, hn_bf);
    k_cvt_bf16<<<16384, 256, 0, stream>>>(head_w, hw_bf, V_ * E_);
    k_gemm_bf16<<<dim3(V_ / 128, TOK / 128), 256, 0, stream>>>(
        hn_bf, hw_bf, head_b, nullptr, (float*)d_out, nullptr, TOK, V_, E_, 0);
}
